// GNN_node_19885698580757
// MI455X (gfx1250) — matrix operand order repaired and checked by reading, not compiled, not run on hardware
//
#include <hip/hip_runtime.h>
#include <math.h>

// ---------------------------------------------------------------------------
// GNN graph-transformer for MI455X (gfx1250, wave32).
// GEMMs: v_wmma_f32_16x16x32_f16; A tile staged via CDNA5 async direct-to-LDS
// (global_load_async_to_lds_b128 + s_wait_asynccnt), B prepacked to fragment
// order (32B contiguous loads). Edge softmax/scatter stays fp32 VALU with
// wave-per-edge coalescing.
// ---------------------------------------------------------------------------

#define Dm 128
#define Hh 4
#define DhH 32
#define NEFF 7
#define LLAY 4

#define USE_ASYNC_COPY 1   // set 0 to fall back to sync VMEM->VGPR->LDS staging

typedef __attribute__((ext_vector_type(16))) _Float16 v16h;
typedef __attribute__((ext_vector_type(8)))  _Float16 h8;
typedef __attribute__((ext_vector_type(8)))  float    v8f;

// -------------------------------- helpers ----------------------------------

__device__ __forceinline__ float wave_sum32(float v) {
    #pragma unroll
    for (int off = 16; off > 0; off >>= 1) v += __shfl_xor(v, off, 32);
    return v;
}

__device__ __forceinline__ void atomicMaxFloat(float* addr, float value) {
    if (value >= 0.0f) {
        atomicMax((int*)addr, __float_as_int(value));
    } else {
        atomicMin((unsigned int*)addr, (unsigned int)__float_as_int(value));
    }
}

__device__ __forceinline__ float leaky(float x) { return x > 0.0f ? x : 0.01f * x; }

// Async 16B global -> LDS copy (CDNA5 TDM-adjacent path, ASYNCcnt-tracked).
__device__ __forceinline__ void async_copy_b128(void* lds_dst, const void* gsrc) {
#if USE_ASYNC_COPY
    const unsigned loff = (unsigned)(uintptr_t)lds_dst;       // LDS address from low 32 bits of pointer
    const unsigned long long ga = (unsigned long long)(uintptr_t)gsrc;
    asm volatile("global_load_async_to_lds_b128 %0, %1, off"
                 :: "v"(loff), "v"(ga) : "memory");
#else
    *(h8*)lds_dst = *(const h8*)gsrc;
#endif
}

__device__ __forceinline__ void async_copy_wait() {
#if USE_ASYNC_COPY
    asm volatile("s_wait_asynccnt 0x0" ::: "memory");
#endif
}

// ------------------------------ tiny kernels --------------------------------

// h[i*D + c] = node_table[c]; also emit f16 copy for first-layer GEMM A-operand
__global__ __launch_bounds__(256) void broadcast_nodes(const float* __restrict__ table,
                                                       float* __restrict__ h,
                                                       _Float16* __restrict__ h16,
                                                       long long n) {
    long long t = (long long)blockIdx.x * 256 + threadIdx.x;
    if (t < n) {
        const float v = table[t & (Dm - 1)];
        h[t] = v;
        h16[t] = (_Float16)v;
    }
}

// agg = 0; mmax = -1e30; den = 0
__global__ __launch_bounds__(256) void attn_init(float* __restrict__ agg,
                                                 float* __restrict__ mmax,
                                                 float* __restrict__ den,
                                                 long long nND, long long nNH) {
    long long t = (long long)blockIdx.x * 256 + threadIdx.x;
    if (t < nND) agg[t] = 0.0f;
    if (t < nNH) { mmax[t] = -1.0e30f; den[t] = 0.0f; }
}

// ee_f16 = edge_attr @ Wee + bee   (K = 7, pure VALU: memory bound)
__global__ __launch_bounds__(256) void edge_encode(const float* __restrict__ ea,
                                                   const float* __restrict__ Wee,
                                                   const float* __restrict__ bee,
                                                   _Float16* __restrict__ ee16,
                                                   long long total) {
    long long t = (long long)blockIdx.x * 256 + threadIdx.x;
    if (t >= total) return;
    int c = (int)(t & (Dm - 1));
    long long e = t >> 7;
    float s = bee[c];
    #pragma unroll
    for (int j = 0; j < NEFF; ++j) s += ea[e * NEFF + j] * Wee[j * Dm + c];
    ee16[t] = (_Float16)s;
}

// Prepack f32 weight W[K][NC] into WMMA-B fragment order (f16):
//   Bp[((nt*(K/32) + ks)*32 + lane)*16 + i] = W[ks*32 + (lane>>4)*16 + i][nt*16 + (lane&15)]
__global__ __launch_bounds__(256) void prepack_b(const float* __restrict__ W,
                                                 _Float16* __restrict__ Bp,
                                                 int K, int NC) {
    long long t = (long long)blockIdx.x * 256 + threadIdx.x;
    if (t >= (long long)K * NC) return;
    const int i    = (int)(t & 15);
    const int lane = (int)((t >> 4) & 31);
    long long rest = t >> 9;
    const int kst  = K >> 5;                 // K/32 k-steps
    const int ks   = (int)(rest % kst);
    const int nt   = (int)(rest / kst);
    const int k = ks * 32 + (lane >> 4) * 16 + i;
    const int n = nt * 16 + (lane & 15);
    Bp[t] = (_Float16)W[(size_t)k * NC + n];
}

// ------------------------------ WMMA GEMM ----------------------------------
// C[M,NC] = A[M,K](f16) @ W(prepacked f16) + bias; epilogue compiled in via
// template flags (f32 out / f16 out / leaky). Block = 256 threads = 8 waves =
// one 16-row M-tile; wave w owns N-tiles {w, w+8}. A tile staged in padded LDS
// through the async direct-to-LDS path.
template <int K, int NC, bool O32, bool O16, bool ACT>
__global__ __launch_bounds__(256) void gemm_wmma(const _Float16* __restrict__ A,
                                                 const _Float16* __restrict__ Bp,
                                                 const float* __restrict__ bias,
                                                 float* __restrict__ out32,
                                                 _Float16* __restrict__ out16,
                                                 int M) {
    constexpr int SP = K + 8;                // padded LDS row stride (f16)
    constexpr int KSTEPS = K / 32;
    constexpr int NTW = NC / 128;            // N-tiles per wave (1 or 2)
    __shared__ _Float16 As[16 * SP];

    const int lane  = threadIdx.x & 31;
    const int wave  = threadIdx.x >> 5;
    const int mtile = blockIdx.x;
    if (mtile * 16 >= M) return;

    // A-tile staging: 16 rows x K f16, one aligned 16B chunk per thread pass
    constexpr int CH = K / 8;                // h8 chunks per row
    #pragma unroll
    for (int i = threadIdx.x; i < 16 * CH; i += 256) {
        const int r = i / CH, c = i % CH;
        async_copy_b128(&As[r * SP + c * 8],
                        A + (size_t)(mtile * 16 + r) * K + c * 8);
    }
    async_copy_wait();
    __syncthreads();

    const int lo = lane & 15;
    const int hi = lane >> 4;

    v8f acc[NTW];
    #pragma unroll
    for (int j = 0; j < NTW; ++j) {
        const float bv = bias[(wave + 8 * j) * 16 + lo];
        #pragma unroll
        for (int r = 0; r < 8; ++r) acc[j][r] = bv;
    }

    #pragma unroll
    for (int ks = 0; ks < KSTEPS; ++ks) {
        // A fragment (16-bit A 16x32 layout): two aligned 16B LDS reads
        const h8 alo = *(const h8*)&As[lo * SP + ks * 32 + hi * 16];
        const h8 ahi = *(const h8*)&As[lo * SP + ks * 32 + hi * 16 + 8];
        v16h a;
        #pragma unroll
        for (int i = 0; i < 8; ++i) { a[i] = alo[i]; a[i + 8] = ahi[i]; }

        #pragma unroll
        for (int j = 0; j < NTW; ++j) {
            const int nt = wave + 8 * j;
            const v16h b = *(const v16h*)(Bp +
                (((size_t)nt * KSTEPS + ks) * 32 + lane) * 16);
            acc[j] = __builtin_amdgcn_wmma_f32_16x16x32_f16(
                false, a, false, b, (short)0, acc[j], false, false);
        }
    }

    #pragma unroll
    for (int j = 0; j < NTW; ++j) {
        const int n_g = (wave + 8 * j) * 16 + lo;
        #pragma unroll
        for (int r = 0; r < 8; ++r) {
            const int mg = mtile * 16 + r + hi * 8;
            float val = acc[j][r];
            if (ACT) val = leaky(val);
            const size_t idx = (size_t)mg * NC + n_g;
            if (O32) out32[idx] = val;
            if (O16) out16[idx] = (_Float16)val;
        }
    }
}

// --------------------------- attention kernels ------------------------------
// Wave per edge: lane l covers column h*32+l of the 128-wide row for h=0..3.

__global__ __launch_bounds__(128) void attn_alpha(const float* __restrict__ q,
                                                  const float* __restrict__ k,
                                                  const float* __restrict__ e,
                                                  const int* __restrict__ src,
                                                  const int* __restrict__ dst,
                                                  float* __restrict__ alpha,
                                                  float* __restrict__ mmax,
                                                  int nE, float scale) {
    const int lane = threadIdx.x & 31;
    const int eid  = blockIdx.x * 4 + (threadIdx.x >> 5);
    if (eid >= nE) return;
    const int s = src[eid], d = dst[eid];
    const float* qp = q + (size_t)d * Dm;
    const float* kp = k + (size_t)s * Dm;
    const float* ep = e + (size_t)eid * Dm;
    float acc[Hh];
    #pragma unroll
    for (int h = 0; h < Hh; ++h) {
        const int c = h * DhH + lane;
        acc[h] = qp[c] * (kp[c] + ep[c]);
    }
    #pragma unroll
    for (int h = 0; h < Hh; ++h) acc[h] = wave_sum32(acc[h]);
    if (lane == 0) {
        #pragma unroll
        for (int h = 0; h < Hh; ++h) {
            const float a = acc[h] * scale;
            alpha[(size_t)eid * Hh + h] = a;
            atomicMaxFloat(&mmax[(size_t)d * Hh + h], a);
        }
    }
}

__global__ __launch_bounds__(256) void attn_exp(const int* __restrict__ dst,
                                                float* __restrict__ alpha,
                                                const float* __restrict__ mmax,
                                                float* __restrict__ den,
                                                long long total) {
    long long t = (long long)blockIdx.x * 256 + threadIdx.x;
    if (t >= total) return;
    const long long eid = t >> 2;
    const int h = (int)(t & 3);
    const int d = dst[eid];
    const float ex = __expf(alpha[t] - mmax[(size_t)d * Hh + h]);
    alpha[t] = ex;
    atomicAdd(&den[(size_t)d * Hh + h], ex);
}

__global__ __launch_bounds__(128) void attn_agg(const float* __restrict__ v,
                                                const float* __restrict__ e,
                                                const int* __restrict__ src,
                                                const int* __restrict__ dst,
                                                const float* __restrict__ alpha,
                                                const float* __restrict__ den,
                                                float* __restrict__ agg,
                                                int nE) {
    const int lane = threadIdx.x & 31;
    const int eid  = blockIdx.x * 4 + (threadIdx.x >> 5);
    if (eid >= nE) return;
    const int s = src[eid], d = dst[eid];
    const float* vp = v + (size_t)s * Dm;
    const float* ep = e + (size_t)eid * Dm;
    #pragma unroll
    for (int h = 0; h < Hh; ++h) {
        const float attn = alpha[(size_t)eid * Hh + h] /
                           (den[(size_t)d * Hh + h] + 1e-16f);
        const int c = h * DhH + lane;
        atomicAdd(&agg[(size_t)d * Dm + c], attn * (vp[c] + ep[c]));
    }
}

// ----------------------- fused LayerNorm kernels ----------------------------
// Wave per row (128 floats = 32 lanes x 4), shuffle reductions, no LDS.

// x1 = LN(h + agg + s)*g + b ; writes f32 + f16
__global__ __launch_bounds__(128) void residual_ln3(const float* __restrict__ hA,
                                                    const float* __restrict__ hB,
                                                    const float* __restrict__ hC,
                                                    const float* __restrict__ g,
                                                    const float* __restrict__ bt,
                                                    float* __restrict__ out32,
                                                    _Float16* __restrict__ out16,
                                                    int rows) {
    const int lane = threadIdx.x & 31;
    const int row  = blockIdx.x * 4 + (threadIdx.x >> 5);
    if (row >= rows) return;
    const size_t base = (size_t)row * Dm;
    float x[4], s = 0.f, s2 = 0.f;
    #pragma unroll
    for (int i = 0; i < 4; ++i) {
        const int c = lane + 32 * i;
        x[i] = hA[base + c] + hB[base + c] + hC[base + c];
        s += x[i]; s2 += x[i] * x[i];
    }
    s = wave_sum32(s); s2 = wave_sum32(s2);
    const float mu  = s * (1.0f / Dm);
    const float inv = rsqrtf(s2 * (1.0f / Dm) - mu * mu + 1e-5f);
    #pragma unroll
    for (int i = 0; i < 4; ++i) {
        const int c = lane + 32 * i;
        const float y = (x[i] - mu) * inv * g[c] + bt[c];
        out32[base + c] = y;
        out16[base + c] = (_Float16)y;
    }
}

// hconv = LN(x1 + ff); hn = LN(hconv); optional leaky; h = hn + h_in (in place)
// Also emits h16 for the next layer's WMMA A-operand.
__global__ __launch_bounds__(128) void double_ln(const float* __restrict__ x1,
                                                 const float* __restrict__ ff,
                                                 const float* __restrict__ g2,
                                                 const float* __restrict__ b2,
                                                 const float* __restrict__ g3,
                                                 const float* __restrict__ b3,
                                                 float* __restrict__ h,  // in/out
                                                 _Float16* __restrict__ h16,
                                                 int act, int rows) {
    const int lane = threadIdx.x & 31;
    const int row  = blockIdx.x * 4 + (threadIdx.x >> 5);
    if (row >= rows) return;
    const size_t base = (size_t)row * Dm;
    float x[4], hin[4], s = 0.f, s2 = 0.f;
    #pragma unroll
    for (int i = 0; i < 4; ++i) {
        const int c = lane + 32 * i;
        hin[i] = h[base + c];
        x[i] = x1[base + c] + ff[base + c];
        s += x[i]; s2 += x[i] * x[i];
    }
    s = wave_sum32(s); s2 = wave_sum32(s2);
    float mu  = s * (1.0f / Dm);
    float inv = rsqrtf(s2 * (1.0f / Dm) - mu * mu + 1e-5f);
    s = 0.f; s2 = 0.f;
    #pragma unroll
    for (int i = 0; i < 4; ++i) {
        const int c = lane + 32 * i;
        x[i] = (x[i] - mu) * inv * g2[c] + b2[c];   // hconv
        s += x[i]; s2 += x[i] * x[i];
    }
    s = wave_sum32(s); s2 = wave_sum32(s2);
    mu  = s * (1.0f / Dm);
    inv = rsqrtf(s2 * (1.0f / Dm) - mu * mu + 1e-5f);
    #pragma unroll
    for (int i = 0; i < 4; ++i) {
        const int c = lane + 32 * i;
        float y = (x[i] - mu) * inv * g3[c] + b3[c];
        if (act) y = leaky(y);
        y += hin[i];
        h[base + c] = y;
        h16[base + c] = (_Float16)y;
    }
}

// ------------------------------- launcher -----------------------------------

extern "C" void kernel_launch(void* const* d_in, const int* in_sizes, int n_in,
                              void* d_out, int out_size, void* d_ws, size_t ws_size,
                              hipStream_t stream) {
    (void)n_in; (void)out_size; (void)ws_size;
    const int*   edge_index = (const int*)d_in[1];
    const float* edge_attr  = (const float*)d_in[2];
    const float* node_table = (const float*)d_in[3];
    const float* Wee = (const float*)d_in[4];
    const float* bee = (const float*)d_in[5];
    const float* Wq = (const float*)d_in[6],  *bq = (const float*)d_in[7];
    const float* Wk = (const float*)d_in[8],  *bk = (const float*)d_in[9];
    const float* Wv = (const float*)d_in[10], *bv = (const float*)d_in[11];
    const float* We = (const float*)d_in[12], *be = (const float*)d_in[13];
    const float* Ws = (const float*)d_in[14], *bs = (const float*)d_in[15];
    const float* W1 = (const float*)d_in[16], *b1 = (const float*)d_in[17];
    const float* W2 = (const float*)d_in[18], *b2 = (const float*)d_in[19];
    const float* n1g = (const float*)d_in[20], *n1b = (const float*)d_in[21];
    const float* n2g = (const float*)d_in[22], *n2b = (const float*)d_in[23];
    const float* lng = (const float*)d_in[24], *lnb = (const float*)d_in[25];

    const int N = in_sizes[0];
    const int E = in_sizes[1] / 2;
    const int* src = edge_index;
    const int* dst = edge_index + E;
    float* h = (float*)d_out;                 // live node state, final answer

    // ---- workspace carve-out (256B aligned) ----
    char* p = (char*)d_ws;
    auto alloc = [&](size_t bytes) -> char* {
        char* r = p; p += (bytes + 255) & ~(size_t)255; return r;
    };
    _Float16* ee16  = (_Float16*)alloc((size_t)E * Dm * 2);
    float*    ebuf  = (float*)   alloc((size_t)E * Dm * 4);
    float*    qb    = (float*)   alloc((size_t)N * Dm * 4);
    float*    kb    = (float*)   alloc((size_t)N * Dm * 4);
    float*    vb    = (float*)   alloc((size_t)N * Dm * 4);
    float*    sb    = (float*)   alloc((size_t)N * Dm * 4);
    float*    agg   = (float*)   alloc((size_t)N * Dm * 4);
    float*    x1f   = (float*)   alloc((size_t)N * Dm * 4);
    float*    ff2   = (float*)   alloc((size_t)N * Dm * 4);
    _Float16* h16   = (_Float16*)alloc((size_t)N * Dm * 2);
    _Float16* x116  = (_Float16*)alloc((size_t)N * Dm * 2);
    _Float16* ff116 = (_Float16*)alloc((size_t)N * 2 * Dm * 2);
    float*    alpha = (float*)   alloc((size_t)E * Hh * 4);
    float*    mmax  = (float*)   alloc((size_t)N * Hh * 4);
    float*    den   = (float*)   alloc((size_t)N * Hh * 4);
    const size_t wSq = (size_t)Dm * Dm;          // per-layer elems, square W
    const size_t wFf = (size_t)Dm * 2 * Dm;      // per-layer elems, FFN W
    _Float16* wq16 = (_Float16*)alloc(LLAY * wSq * 2);
    _Float16* wk16 = (_Float16*)alloc(LLAY * wSq * 2);
    _Float16* wv16 = (_Float16*)alloc(LLAY * wSq * 2);
    _Float16* we16 = (_Float16*)alloc(LLAY * wSq * 2);
    _Float16* ws16 = (_Float16*)alloc(LLAY * wSq * 2);
    _Float16* w116 = (_Float16*)alloc(LLAY * wFf * 2);
    _Float16* w216 = (_Float16*)alloc(LLAY * wFf * 2);

    const long long nND = (long long)N * Dm;
    const long long nNH = (long long)N * Hh;
    const long long nED = (long long)E * Dm;
    const long long nEH = (long long)E * Hh;
    const float scale = 0.17677669529663687f;   // 1/sqrt(32)

    auto blk = [](long long n, int b) { return (unsigned)((n + b - 1) / b); };

    // ---- one-time prep: prepack all weights into WMMA fragment order ----
    for (int l = 0; l < LLAY; ++l) {
        prepack_b<<<blk(wSq, 256), 256, 0, stream>>>(Wq + l * wSq, wq16 + l * wSq, Dm, Dm);
        prepack_b<<<blk(wSq, 256), 256, 0, stream>>>(Wk + l * wSq, wk16 + l * wSq, Dm, Dm);
        prepack_b<<<blk(wSq, 256), 256, 0, stream>>>(Wv + l * wSq, wv16 + l * wSq, Dm, Dm);
        prepack_b<<<blk(wSq, 256), 256, 0, stream>>>(We + l * wSq, we16 + l * wSq, Dm, Dm);
        prepack_b<<<blk(wSq, 256), 256, 0, stream>>>(Ws + l * wSq, ws16 + l * wSq, Dm, Dm);
        prepack_b<<<blk(wFf, 256), 256, 0, stream>>>(W1 + l * wFf, w116 + l * wFf, Dm, 2 * Dm);
        prepack_b<<<blk(wFf, 256), 256, 0, stream>>>(W2 + l * wFf, w216 + l * wFf, 2 * Dm, Dm);
    }
    broadcast_nodes<<<blk(nND, 256), 256, 0, stream>>>(node_table, h, h16, nND);
    edge_encode<<<blk(nED, 256), 256, 0, stream>>>(edge_attr, Wee, bee, ee16, nED);

    const unsigned gN = (unsigned)(N / 16);
    const unsigned gE = (unsigned)(E / 16);
    const unsigned edgeWaves = blk(E, 4);   // 4 waves / 128-thread block
    const unsigned rowWaves  = blk(N, 4);

    for (int l = 0; l < LLAY; ++l) {
        const size_t oS = (size_t)l * wSq;
        const size_t oF = (size_t)l * wFf;
        // q, k, v, skip projections + edge projection (all WMMA)
        gemm_wmma<Dm, Dm, true, false, false><<<gN, 256, 0, stream>>>(
            h16, wq16 + oS, bq + l * Dm, qb, nullptr, N);
        gemm_wmma<Dm, Dm, true, false, false><<<gN, 256, 0, stream>>>(
            h16, wk16 + oS, bk + l * Dm, kb, nullptr, N);
        gemm_wmma<Dm, Dm, true, false, false><<<gN, 256, 0, stream>>>(
            h16, wv16 + oS, bv + l * Dm, vb, nullptr, N);
        gemm_wmma<Dm, Dm, true, false, false><<<gN, 256, 0, stream>>>(
            h16, ws16 + oS, bs + l * Dm, sb, nullptr, N);
        gemm_wmma<Dm, Dm, true, false, false><<<gE, 256, 0, stream>>>(
            ee16, we16 + oS, be + l * Dm, ebuf, nullptr, E);
        // edge softmax + scatter aggregate
        attn_init<<<blk(nND, 256), 256, 0, stream>>>(agg, mmax, den, nND, nNH);
        attn_alpha<<<edgeWaves, 128, 0, stream>>>(qb, kb, ebuf, src, dst, alpha, mmax, E, scale);
        attn_exp<<<blk(nEH, 256), 256, 0, stream>>>(dst, alpha, mmax, den, nEH);
        attn_agg<<<edgeWaves, 128, 0, stream>>>(vb, ebuf, src, dst, alpha, den, agg, E);
        // x1 = LN(h + agg + skip)
        residual_ln3<<<rowWaves, 128, 0, stream>>>(h, agg, sb, n1g + l * Dm, n1b + l * Dm,
                                                   x1f, x116, N);
        // FFN: leaky(x1 @ W1 + b1) @ W2 + b2
        gemm_wmma<Dm, 2 * Dm, false, true, true><<<gN, 256, 0, stream>>>(
            x116, w116 + oF, b1 + l * 2 * Dm, nullptr, ff116, N);
        gemm_wmma<2 * Dm, Dm, true, false, false><<<gN, 256, 0, stream>>>(
            ff116, w216 + oF, b2 + l * Dm, ff2, nullptr, N);
        // hconv = LN(x1+ff); hn = LN(hconv); act (l<3); h = hn + h_in; emit h16
        double_ln<<<rowWaves, 128, 0, stream>>>(x1f, ff2, n2g + l * Dm, n2b + l * Dm,
                                                lng + l * Dm, lnb + l * Dm, h, h16,
                                                (l < LLAY - 1) ? 1 : 0, N);
    }
}
